// FRCNN_55645596287810
// MI455X (gfx1250) — matrix operand order repaired and checked by reading, not compiled, not run on hardware
//
#include <hip/hip_runtime.h>
#include <hip/hip_bf16.h>
#include <math.h>

// ---------------- types for CDNA5 WMMA ----------------
typedef __attribute__((ext_vector_type(16))) __bf16          v16bf;
typedef __attribute__((ext_vector_type(16))) unsigned short  v16us;
typedef __attribute__((ext_vector_type(8)))  float           v8f;

// pack two fp32 -> two bf16 (round-to-nearest) in one dword:
// two v_add + one v_perm_b32
__device__ __forceinline__ unsigned int f2bf2(float lo, float hi) {
  unsigned int ul = __builtin_bit_cast(unsigned int, lo) + 0x8000u;
  unsigned int uh = __builtin_bit_cast(unsigned int, hi) + 0x8000u;
  return __builtin_amdgcn_perm(uh, ul, 0x07060302u);  // {uh[3],uh[2],ul[3],ul[2]}
}

// ---------------- constants ----------------
#define FH 50
#define FW 50
#define NPOS (FH*FW)          // 2500
#define CIN 512
#define K3 (CIN*9)            // 4608
#define NA (NPOS*9)           // 22500 anchors
#define NSORT 32768
#define PRE_NMS 12000
#define N_SAMPLE 128
#define ROI_SIZE 7
#define KMAX 10
#define NMS_IOU 0.7f
#define MIN_SZ 0.016f
#define FC1_K (CIN*ROI_SIZE*ROI_SIZE)   // 25088
#define FC_N 4096
#define NCLS 21

// =====================================================================
// GEMM: C[M][N] = A[M][K] * B[K][N]  (fp32 in, bf16 WMMA, fp32 out)
// Block tile 64x128, 8 waves in 2x4 grid, each wave = 32x32 C block
// = 4 x v_wmma_f32_16x16x32_bf16 per K-step.
// Software-pipelined: global loads for tile i+1 issue before the barrier
// and overlap the WMMA phase of tile i; LDS double buffer -> ONE barrier
// per K-step.
// REQUIRES K % 32 == 0 (true for all launches here).
// M/N edges handled by index clamping: OOB tile elements feed only
// C rows/cols that the store phase discards, so no value masking needed.
// biasMode: 0 none, 1 bias[M], 2 bias[N]
// =====================================================================
#define TM 64
#define TN 128
#define TK 32

__global__ __launch_bounds__(256)
void gemm_bf16_wmma(const float* __restrict__ A, const float* __restrict__ B,
                    float* __restrict__ C, const float* __restrict__ bias,
                    int M, int N, int K, int biasMode, int relu)
{
  __shared__ unsigned short As[2][TM][TK];   // 2 x 4 KB
  __shared__ unsigned short Bs[2][TN][TK];   // 2 x 8 KB

  const int tid  = threadIdx.x;
  const int lane = tid & 31;
  const int wave = tid >> 5;       // 0..7
  const int wmW  = wave >> 2;      // 0..1 : 32-row band
  const int wnW  = wave & 3;       // 0..3 : 32-col band
  const int bm   = blockIdx.y * TM;
  const int bn   = blockIdx.x * TN;

  const int lm = lane & 15;
  const int ah = (lane >> 4) * 8;      // A half K-offset
  const int bh = (lane >> 4) * 16;     // unused (B fragments use ah)

  // ---- A loader mapping: thread -> (row, 8-wide k segment) ----
  const int am  = tid >> 2;            // 0..63
  const int as8 = (tid & 3) * 8;       // 0,8,16,24
  int agm = bm + am; agm = (agm < M) ? agm : (M - 1);      // clamp, hoisted

  // ---- B loader mapping: thread -> (col, 16-wide k segment) ----
  const int bnn = tid >> 1;            // 0..127
  const int bks = (tid & 1) * 16;      // 0 or 16
  int gnc = bn + bnn; gnc = (gnc < N) ? gnc : (N - 1);     // clamp, hoisted

  const float* Abase = A + (size_t)agm * K + as8;   // 16B aligned (K%32==0)
  const float* Bbase = B + gnc;

  float4 ra0, ra1;
  float  rb[16];
  auto load_tile = [&](int kk) {
    const float4* Ap = (const float4*)(Abase + kk);
    ra0 = Ap[0];
    ra1 = Ap[1];
    const float* Bp = Bbase + (size_t)(kk + bks) * N;
    #pragma unroll
    for (int j = 0; j < 16; ++j) rb[j] = Bp[(size_t)j * N];
  };

  v8f acc[2][2] = {};
  load_tile(0);
  int cur = 0;

  for (int k0 = 0; k0 < K; k0 += TK) {
    // ---- pack previously-loaded tile into LDS[cur] ----
    uint4 pa;
    pa.x = f2bf2(ra0.x, ra0.y);  pa.y = f2bf2(ra0.z, ra0.w);
    pa.z = f2bf2(ra1.x, ra1.y);  pa.w = f2bf2(ra1.z, ra1.w);
    *(uint4*)&As[cur][am][as8] = pa;
    uint4 p0, p1;
    p0.x = f2bf2(rb[0],  rb[1]);   p0.y = f2bf2(rb[2],  rb[3]);
    p0.z = f2bf2(rb[4],  rb[5]);   p0.w = f2bf2(rb[6],  rb[7]);
    p1.x = f2bf2(rb[8],  rb[9]);   p1.y = f2bf2(rb[10], rb[11]);
    p1.z = f2bf2(rb[12], rb[13]);  p1.w = f2bf2(rb[14], rb[15]);
    *(uint4*)&Bs[cur][bnn][bks]     = p0;
    *(uint4*)&Bs[cur][bnn][bks + 8] = p1;

    // ---- issue NEXT tile's global loads early (overlap with WMMA) ----
    if (k0 + TK < K) load_tile(k0 + TK);

    __syncthreads();   // single barrier per K-step (double buffer)

    // ---- fragments per ISA VGPR layout (vectorizes to ds_load_b128) ----
    v16us au[2], bu[2];
    #pragma unroll
    for (int sm = 0; sm < 2; ++sm) {
      int mrow = wmW * 32 + sm * 16 + lm;
      #pragma unroll
      for (int e = 0; e < 16; ++e)
        au[sm][e] = As[cur][mrow][((e & 8) << 1) + (e & 7) + ah];
    }
    #pragma unroll
    for (int sn = 0; sn < 2; ++sn) {
      int nrow = wnW * 32 + sn * 16 + lm;
      #pragma unroll
      for (int e = 0; e < 16; ++e)
        bu[sn][e] = Bs[cur][nrow][((e & 8) << 1) + (e & 7) + ah];
    }
    #pragma unroll
    for (int sm = 0; sm < 2; ++sm) {
      #pragma unroll
      for (int sn = 0; sn < 2; ++sn) {
        acc[sm][sn] = __builtin_amdgcn_wmma_f32_16x16x32_bf16(
            false, __builtin_bit_cast(v16bf, au[sm]),
            false, __builtin_bit_cast(v16bf, bu[sn]),
            (short)0, acc[sm][sn], false, false);
      }
    }
    cur ^= 1;
  }

  // ---- store D (lane L: M = r + (L/16)*8, N = L%16) ----
  #pragma unroll
  for (int sm = 0; sm < 2; ++sm) {
    #pragma unroll
    for (int sn = 0; sn < 2; ++sn) {
      #pragma unroll
      for (int r = 0; r < 8; ++r) {
        int gm = bm + wmW * 32 + sm * 16 + r + ((lane >> 4) << 3);
        int gn = bn + wnW * 32 + sn * 16 + lm;
        if (gm < M && gn < N) {
          float v = acc[sm][sn][r];
          if (biasMode == 1)      v += bias[gm];
          else if (biasMode == 2) v += bias[gn];
          if (relu) v = fmaxf(v, 0.0f);
          C[(size_t)gm * N + gn] = v;
        }
      }
    }
  }
}

// =====================================================================
// im2col for 3x3 SAME conv : col[k][pos], k = i*9 + r*3 + s
// =====================================================================
__global__ void im2col_k(const float* __restrict__ f, float* __restrict__ col)
{
  int idx = blockIdx.x * blockDim.x + threadIdx.x;
  if (idx >= K3 * NPOS) return;
  int pos = idx % NPOS;
  int k   = idx / NPOS;
  int s = k % 3, r = (k / 3) % 3, i = k / 9;
  int x = pos % FW, y = pos / FW;
  int yy = y + r - 1, xx = x + s - 1;
  float v = 0.0f;
  if (yy >= 0 && yy < FH && xx >= 0 && xx < FW) v = f[(i * FH + yy) * FW + xx];
  col[idx] = v;
}

// =====================================================================
// scatter NCHW conv outputs -> NHWC-flattened RPN outputs in d_out
// =====================================================================
__global__ void rpn_out_k(const float* __restrict__ Ccls,
                          const float* __restrict__ Creg,
                          float* __restrict__ out)
{
  int t = blockIdx.x * blockDim.x + threadIdx.x;
  if (t >= NPOS * 54) return;
  int ch = t % 54, pos = t / 54;
  if (ch < 18) out[pos * 18 + ch]                 = Ccls[ch * NPOS + pos];
  else         out[NA * 2 + pos * 36 + (ch - 18)] = Creg[(ch - 18) * NPOS + pos];
}

// =====================================================================
// softmax-fg, anchor decode, clip, min-size filter -> scores/boxes
// =====================================================================
__global__ void decode_k(const float* __restrict__ Ccls,
                         const float* __restrict__ Creg,
                         const float* __restrict__ anchor,
                         float* __restrict__ boxes,
                         float* __restrict__ score,
                         int*   __restrict__ sidx)
{
  int a = blockIdx.x * blockDim.x + threadIdx.x;
  if (a >= NSORT) return;
  sidx[a] = a;
  if (a >= NA) { score[a] = -INFINITY; return; }
  int pos = a / 9, kk = a % 9;

  float l0 = Ccls[(kk * 2 + 0) * NPOS + pos];
  float l1 = Ccls[(kk * 2 + 1) * NPOS + pos];
  float fg = 1.0f / (1.0f + expf(l0 - l1));

  float gx = Creg[(kk * 4 + 0) * NPOS + pos];
  float gy = Creg[(kk * 4 + 1) * NPOS + pos];
  float gw = Creg[(kk * 4 + 2) * NPOS + pos];
  float gh = Creg[(kk * 4 + 3) * NPOS + pos];

  float a0 = anchor[a * 4 + 0], a1 = anchor[a * 4 + 1];
  float a2 = anchor[a * 4 + 2], a3 = anchor[a * 4 + 3];
  float cx = 0.5f * (a0 + a2), cy = 0.5f * (a1 + a3);
  float w  = a2 - a0,          h  = a3 - a1;
  float ncx = gx * w + cx, ncy = gy * h + cy;
  float nw  = expf(gw) * w, nh = expf(gh) * h;

  float x1 = fminf(fmaxf(ncx - 0.5f * nw, 0.0f), 1.0f);
  float y1 = fminf(fmaxf(ncy - 0.5f * nh, 0.0f), 1.0f);
  float x2 = fminf(fmaxf(ncx + 0.5f * nw, 0.0f), 1.0f);
  float y2 = fminf(fmaxf(ncy + 0.5f * nh, 0.0f), 1.0f);

  boxes[a * 4 + 0] = x1; boxes[a * 4 + 1] = y1;
  boxes[a * 4 + 2] = x2; boxes[a * 4 + 3] = y2;

  bool keep = ((y2 - y1) >= MIN_SZ) && ((x2 - x1) >= MIN_SZ);
  score[a] = keep ? fg : -INFINITY;
}

// =====================================================================
// global-memory bitonic sort step (descending), key=score payload=index
// =====================================================================
__global__ void bitonic_step(float* __restrict__ key, int* __restrict__ idx,
                             int j, int k)
{
  int i = blockIdx.x * blockDim.x + threadIdx.x;
  if (i >= NSORT) return;
  int ixj = i ^ j;
  if (ixj <= i) return;
  bool descRegion = ((i & k) == 0);
  float ki = key[i], kj = key[ixj];
  bool doSwap = descRegion ? (ki < kj) : (ki > kj);
  if (doSwap) {
    key[i] = kj; key[ixj] = ki;
    int ti = idx[i]; idx[i] = idx[ixj]; idx[ixj] = ti;
  }
}

// =====================================================================
// gather top-PRE_NMS sorted boxes, areas, initial keep flags
// =====================================================================
__global__ void gather_sorted_k(const float* __restrict__ skey,
                                const int*   __restrict__ sidx,
                                const float* __restrict__ boxes,
                                float* __restrict__ sbox,
                                float* __restrict__ area,
                                int*   __restrict__ keep)
{
  int t = blockIdx.x * blockDim.x + threadIdx.x;
  if (t >= PRE_NMS) return;
  int si = sidx[t];
  float x1 = 0, y1 = 0, x2 = 0, y2 = 0;
  if (si < NA) {
    x1 = boxes[si * 4 + 0]; y1 = boxes[si * 4 + 1];
    x2 = boxes[si * 4 + 2]; y2 = boxes[si * 4 + 3];
  }
  sbox[t * 4 + 0] = x1; sbox[t * 4 + 1] = y1;
  sbox[t * 4 + 2] = x2; sbox[t * 4 + 3] = y2;
  area[t] = (x2 - x1) * (y2 - y1);
  float v = skey[t];
  keep[t] = (v > -INFINITY) ? 1 : 0;   // isfinite(vals)
}

// =====================================================================
// sequential greedy NMS, single workgroup (1024 threads, boxes in regs)
// =====================================================================
__global__ __launch_bounds__(1024)
void nms_k(const float* __restrict__ sbox, const float* __restrict__ area,
           int* __restrict__ keep)
{
  const int tid = threadIdx.x;
  const int PER = (PRE_NMS + 1023) / 1024;   // 12
  float bx0[PER], by0[PER], bx1[PER], by1[PER], ar[PER];
  #pragma unroll
  for (int p = 0; p < PER; ++p) {
    int j = tid + p * 1024;
    if (j < PRE_NMS) {
      bx0[p] = sbox[j * 4 + 0]; by0[p] = sbox[j * 4 + 1];
      bx1[p] = sbox[j * 4 + 2]; by1[p] = sbox[j * 4 + 3];
      ar[p]  = area[j];
    }
  }
  for (int i = 0; i < PRE_NMS; ++i) {
    int ki = keep[i];                     // uniform read
    if (ki) {
      float ix0 = sbox[i * 4 + 0], iy0 = sbox[i * 4 + 1];
      float ix1 = sbox[i * 4 + 2], iy1 = sbox[i * 4 + 3];
      float ia  = area[i];
      #pragma unroll
      for (int p = 0; p < PER; ++p) {
        int j = tid + p * 1024;
        if (j < PRE_NMS && j > i) {
          float iw = fmaxf(fminf(ix1, bx1[p]) - fmaxf(ix0, bx0[p]), 0.0f);
          float ih = fmaxf(fminf(iy1, by1[p]) - fmaxf(iy0, by0[p]), 0.0f);
          float inter = iw * ih;
          float iou = inter / (ia + ar[p] - inter + 1e-12f);
          if (iou > NMS_IOU) keep[j] = 0;   // unique owner write
        }
      }
    }
    __threadfence();
    __syncthreads();
  }
}

// =====================================================================
// collect first 128 rois (kept first, then suppressed in order), scale x50
// =====================================================================
__global__ void collect_rois_k(const float* __restrict__ sbox,
                               const int* __restrict__ keep,
                               float* __restrict__ rois)
{
  if (threadIdx.x != 0 || blockIdx.x != 0) return;
  int c = 0;
  for (int pass = 0; pass < 2 && c < N_SAMPLE; ++pass)
    for (int i = 0; i < PRE_NMS && c < N_SAMPLE; ++i)
      if ((pass == 0) == (keep[i] != 0)) {
        rois[c * 4 + 0] = sbox[i * 4 + 0] * (float)FW;
        rois[c * 4 + 1] = sbox[i * 4 + 1] * (float)FH;
        rois[c * 4 + 2] = sbox[i * 4 + 2] * (float)FW;
        rois[c * 4 + 3] = sbox[i * 4 + 3] * (float)FH;
        ++c;
      }
}

// =====================================================================
// ROI max-pool: block = (roi, bin), threads over channels
// =====================================================================
__global__ __launch_bounds__(256)
void roipool_k(const float* __restrict__ feat, const float* __restrict__ rois,
               float* __restrict__ pool)
{
  int bin = blockIdx.x % (ROI_SIZE * ROI_SIZE);
  int n   = blockIdx.x / (ROI_SIZE * ROI_SIZE);
  int ph = bin / ROI_SIZE, pw = bin % ROI_SIZE;

  int rs0 = (int)lrintf(rois[n * 4 + 0]);
  int rs1 = (int)lrintf(rois[n * 4 + 1]);
  int rs2 = (int)lrintf(rois[n * 4 + 2]);
  int rs3 = (int)lrintf(rois[n * 4 + 3]);
  float rw = fmaxf((float)(rs2 - rs0 + 1), 1.0f);
  float rh = fmaxf((float)(rs3 - rs1 + 1), 1.0f);

  int h0 = min(max((int)floorf((float)ph * rh / ROI_SIZE) + rs1, 0), FH);
  int h1 = min(max((int)ceilf((float)(ph + 1) * rh / ROI_SIZE) + rs1, 0), FH);
  int w0 = min(max((int)floorf((float)pw * rw / ROI_SIZE) + rs0, 0), FW);
  int w1 = min(max((int)ceilf((float)(pw + 1) * rw / ROI_SIZE) + rs0, 0), FW);

  for (int c = threadIdx.x; c < CIN; c += 256) {
    float m = -INFINITY;
    for (int kh = 0; kh < KMAX; ++kh) {
      int hh = h0 + kh;
      if (hh >= h1) break;
      hh = min(hh, FH - 1);
      for (int kw = 0; kw < KMAX; ++kw) {
        int ww = w0 + kw;
        if (ww >= w1) break;
        ww = min(ww, FW - 1);
        m = fmaxf(m, feat[(c * FH + hh) * FW + ww]);
      }
    }
    pool[(size_t)n * FC1_K + c * (ROI_SIZE * ROI_SIZE) + ph * ROI_SIZE + pw] =
        (m > -3.0e38f) ? m : 0.0f;
  }
}

// =====================================================================
// host side
// =====================================================================
extern "C" void kernel_launch(void* const* d_in, const int* in_sizes, int n_in,
                              void* d_out, int out_size, void* d_ws, size_t ws_size,
                              hipStream_t stream)
{
  const float* features = (const float*)d_in[0];
  const float* anchor   = (const float*)d_in[1];
  const float* w_inter  = (const float*)d_in[2];
  const float* b_inter  = (const float*)d_in[3];
  const float* w_cls    = (const float*)d_in[4];
  const float* b_cls    = (const float*)d_in[5];
  const float* w_reg    = (const float*)d_in[6];
  const float* b_reg    = (const float*)d_in[7];
  const float* w_fc1    = (const float*)d_in[8];
  const float* b_fc1    = (const float*)d_in[9];
  const float* w_fc2    = (const float*)d_in[10];
  const float* b_fc2    = (const float*)d_in[11];
  const float* w_ch     = (const float*)d_in[12];
  const float* b_ch     = (const float*)d_in[13];
  const float* w_rh     = (const float*)d_in[14];
  const float* b_rh     = (const float*)d_in[15];
  float* out = (float*)d_out;

  // ---- workspace layout (floats) ----
  float* W = (float*)d_ws;
  float* colP   = W;                               // 11,520,000 (dead after conv)
  float* poolP  = colP;                            // 3,211,264 aliases colP
  size_t o = (size_t)K3 * NPOS;
  float* xP     = W + o;  o += (size_t)CIN * NPOS; // conv3x3 output 512x2500
  float* CclsP  = W + o;  o += 18 * NPOS;
  float* CregP  = W + o;  o += 36 * NPOS;
  float* boxesP = W + o;  o += (size_t)NA * 4;
  float* scoreP = W + o;  o += NSORT;
  int*   sidxP  = (int*)(W + o); o += NSORT;
  float* sboxP  = W + o;  o += (size_t)PRE_NMS * 4;
  float* areaP  = W + o;  o += PRE_NMS;
  int*   keepP  = (int*)(W + o); o += PRE_NMS;
  float* roisP  = W + o;  o += N_SAMPLE * 4;
  float* xf1P   = W + o;  o += (size_t)N_SAMPLE * FC_N;
  float* xf2P   = W + o;  o += (size_t)N_SAMPLE * FC_N;

  // ---- 1) im2col ----
  im2col_k<<<(K3 * NPOS + 255) / 256, 256, 0, stream>>>(features, colP);

  // ---- 2) conv3x3 + ReLU as GEMM: x[512][2500] = W[512][4608] * col ----
  gemm_bf16_wmma<<<dim3((NPOS + TN - 1) / TN, (CIN + TM - 1) / TM), 256, 0, stream>>>(
      w_inter, colP, xP, b_inter, CIN, NPOS, K3, /*bias[M]*/1, /*relu*/1);

  // ---- 3) 1x1 convs ----
  gemm_bf16_wmma<<<dim3((NPOS + TN - 1) / TN, 1), 256, 0, stream>>>(
      w_cls, xP, CclsP, b_cls, 18, NPOS, CIN, 1, 0);
  gemm_bf16_wmma<<<dim3((NPOS + TN - 1) / TN, 1), 256, 0, stream>>>(
      w_reg, xP, CregP, b_reg, 36, NPOS, CIN, 1, 0);

  // ---- 4) RPN outputs into d_out (NHWC flatten) ----
  rpn_out_k<<<(NPOS * 54 + 255) / 256, 256, 0, stream>>>(CclsP, CregP, out);

  // ---- 5) decode + scores ----
  decode_k<<<NSORT / 256, 256, 0, stream>>>(CclsP, CregP, anchor, boxesP, scoreP, sidxP);

  // ---- 6) descending bitonic sort (key,idx) over 32768 ----
  for (int k = 2; k <= NSORT; k <<= 1)
    for (int j = k >> 1; j > 0; j >>= 1)
      bitonic_step<<<NSORT / 256, 256, 0, stream>>>(scoreP, sidxP, j, k);

  // ---- 7) top-12000 gather + NMS + roi collection ----
  gather_sorted_k<<<(PRE_NMS + 255) / 256, 256, 0, stream>>>(
      scoreP, sidxP, boxesP, sboxP, areaP, keepP);
  nms_k<<<1, 1024, 0, stream>>>(sboxP, areaP, keepP);
  collect_rois_k<<<1, 32, 0, stream>>>(sboxP, keepP, roisP);

  // ---- 8) ROI max-pool over the ORIGINAL features ----
  roipool_k<<<N_SAMPLE * ROI_SIZE * ROI_SIZE, 256, 0, stream>>>(features, roisP, poolP);

  // ---- 9) FC stack (WMMA) ----
  gemm_bf16_wmma<<<dim3(FC_N / TN, (N_SAMPLE + TM - 1) / TM), 256, 0, stream>>>(
      poolP, w_fc1, xf1P, b_fc1, N_SAMPLE, FC_N, FC1_K, /*bias[N]*/2, 1);
  gemm_bf16_wmma<<<dim3(FC_N / TN, (N_SAMPLE + TM - 1) / TM), 256, 0, stream>>>(
      xf1P, w_fc2, xf2P, b_fc2, N_SAMPLE, FC_N, FC_N, 2, 1);

  float* out_ch = out + NA * 2 + NA * 4;                 // 135000
  float* out_rh = out_ch + N_SAMPLE * NCLS;              // +2688
  gemm_bf16_wmma<<<dim3((NCLS + TN - 1) / TN, (N_SAMPLE + TM - 1) / TM), 256, 0, stream>>>(
      xf2P, w_ch, out_ch, b_ch, N_SAMPLE, NCLS, FC_N, 2, 0);
  gemm_bf16_wmma<<<dim3((NCLS * 4 + TN - 1) / TN, (N_SAMPLE + TM - 1) / TM), 256, 0, stream>>>(
      xf2P, w_rh, out_rh, b_rh, N_SAMPLE, NCLS * 4, FC_N, 2, 0);
}
